// SelfAttention_73890617361023
// MI455X (gfx1250) — matrix operand order repaired and checked by reading, not compiled, not run on hardware
//
#include <hip/hip_runtime.h>

// Self-attention, B=4 S=2048 D=1024, f16 WMMA pipeline for gfx1250 (wave32).
// Compute-bound (~120 GFLOP); all intermediates (~102 MB) fit MI455X's 192 MB L2,
// so a non-fused multipass pipeline pays almost no HBM tax.
// GEMM core: block = 4 waves, each wave owns a 16x128 output tile; the shared
// 128xK B-chunk is double-buffered in LDS via async global->LDS loads
// (ASYNCcnt, inline asm), overlapping the WMMA stream on the current buffer.

#define EMBED 1024
#define SEQ   2048
#define BATCH 4
#define NROWS (BATCH * SEQ)   // 8192

#define KSTEP  64             // K elements staged per LDS buffer
#define ROWPAD 72             // f16 per LDS row (64 data + 8 pad -> 144B stride)
#define CHUNKROWS 128         // N rows per staged chunk (= blockDim.x)

typedef __attribute__((ext_vector_type(16))) _Float16 v16h;
typedef __attribute__((ext_vector_type(8)))  _Float16 v8h;
typedef __attribute__((ext_vector_type(8)))  float    v8f;

// ---- WMMA helpers (layouts per CDNA5 ISA 7.12.2, wave32) --------------------

static __device__ __forceinline__ v8f wmma16x16x32(v16h a, v16h b, v8f c) {
    // D = A(16x32 f16) x B(32x16 f16) + C(16x16 f32)
    return __builtin_amdgcn_wmma_f32_16x16x32_f16(
        /*neg_a=*/false, a, /*neg_b=*/false, b,
        /*c_mod=*/(short)0, c, /*reuse_a=*/false, /*reuse_b=*/false);
}

// A-matrix 16x32 f16 from row-major memory. `tile` points at (row 0, k0).
// lane L: M = L%16; elems 0..7 = K kb..kb+7, elems 8..15 = K 16+kb..23+kb,
// kb = (L<16) ? 0 : 8.
static __device__ __forceinline__ v16h ld_a_tile(const _Float16* tile, int ld, int lane) {
    int m  = lane & 15;
    int kb = (lane < 16) ? 0 : 8;
    const _Float16* p = tile + (size_t)m * ld + kb;
    v8h lo = *(const v8h*)p;
    v8h hi = *(const v8h*)(p + 16);
    v16h r;
#pragma unroll
    for (int i = 0; i < 8; ++i) { r[i] = lo[i]; r[i + 8] = hi[i]; }
    return r;
}

// B-matrix 32x16 f16 from the LDS-staged chunk (chunk row n holds B^T row n0+n,
// KSTEP contiguous f16, row stride ROWPAD). lane L: n = L%16, kb = (L<16)?0:8.
static __device__ __forceinline__ v16h ld_bt_lds(const _Float16* buf, int row0, int kk, int lane) {
    int n  = lane & 15;
    int kb = (lane < 16) ? 0 : 8;
    const _Float16* p = buf + (row0 + n) * ROWPAD + kk + kb;
    v8h lo = *(const v8h*)p;
    v8h hi = *(const v8h*)(p + 16);
    v16h r;
#pragma unroll
    for (int i = 0; i < 8; ++i) { r[i] = lo[i]; r[i + 8] = hi[i]; }
    return r;
}

// ---- Async global -> LDS staging (ASYNCcnt-tracked, inline asm) ------------

// Copy KSTEP f16 (128B) of one chunk row. Each of the 128 threads owns one row.
// VDST = raw 32-bit LDS byte address (low 32 bits of the flat shared address,
// per the ISA LDS aperture mapping), VADDR = 64-bit global address.
static __device__ __forceinline__ void stage_row(const _Float16* src, _Float16* dst) {
#pragma unroll
    for (int i = 0; i < KSTEP / 8; ++i) {
        unsigned ldsaddr = (unsigned)(size_t)(dst + i * 8);
        unsigned long long gaddr = (unsigned long long)(size_t)(src + i * 8);
        asm volatile("global_load_async_to_lds_b128 %0, %1, off"
                     :
                     : "v"(ldsaddr), "v"(gaddr)
                     : "memory");
    }
}

static __device__ __forceinline__ void wait_stage() {
#if __has_builtin(__builtin_amdgcn_s_wait_asynccnt)
    __builtin_amdgcn_s_wait_asynccnt(0);
#else
    asm volatile("s_wait_asynccnt 0x0" ::: "memory");
#endif
}

// ---- Shared GEMM core -------------------------------------------------------
// One wave computes a 16x128 f32 tile: acc[j] = A(16xK) x Bt(rows n0+16j..).
// The 128x KSTEP B-chunk (shared by all 4 waves) is ping-pong staged in LDS.
static __device__ __forceinline__ void gemm_core(
    const _Float16* __restrict__ aBase, int aLd,     // wave's A tile (row m0, col 0)
    const _Float16* __restrict__ btChunk, int btLd,  // chunk base (row n0, col 0)
    int kLen, _Float16* lds, v8f acc[8], int lane) {
    const int tid = threadIdx.x;
    _Float16* buf0 = lds;
    _Float16* buf1 = lds + CHUNKROWS * ROWPAD;
    const _Float16* myRow = btChunk + (size_t)tid * btLd;
    _Float16* myDst0 = buf0 + tid * ROWPAD;
    _Float16* myDst1 = buf1 + tid * ROWPAD;

    stage_row(myRow, myDst0);
    wait_stage();
    __syncthreads();

    _Float16* cur = buf0;
    int phase = 0;
    for (int k0 = 0; k0 < kLen; k0 += KSTEP) {
        int kn = k0 + KSTEP;
        if (kn < kLen)                       // prefetch next chunk into other buffer
            stage_row(myRow + kn, phase ? myDst0 : myDst1);
#pragma unroll
        for (int kk = 0; kk < KSTEP; kk += 32) {
            v16h a = ld_a_tile(aBase + k0 + kk, aLd, lane);
            v16h bt[8];
#pragma unroll
            for (int j = 0; j < 8; ++j) bt[j] = ld_bt_lds(cur, j * 16, kk, lane);
#pragma unroll
            for (int j = 0; j < 8; ++j) acc[j] = wmma16x16x32(a, bt[j], acc[j]);
        }
        wait_stage();                         // my async writes to 'next' landed
        __syncthreads();                      // whole block done reading 'cur'
        phase ^= 1;
        cur = phase ? buf1 : buf0;
    }
}

// ---- Pass 1: dtype conversion / weight transpose ---------------------------

__global__ void cvt_x_kernel(const float* __restrict__ x, _Float16* __restrict__ xh, int n) {
    int i = blockIdx.x * blockDim.x + threadIdx.x;
    if (i < n) xh[i] = (_Float16)x[i];
}

// wt[e*D + d] = (f16) w[d*D + e]  (W stored [in,out] -> Wt stored [out,in])
__global__ void transpose_w_kernel(const float* __restrict__ w, _Float16* __restrict__ wt) {
    int o = blockIdx.x * blockDim.x + threadIdx.x;     // o = e*D + d
    int e = o >> 10;
    int d = o & 1023;
    wt[o] = (_Float16)w[(size_t)d * EMBED + e];
}

// ---- Pass 2: QKV projections ------------------------------------------------
// vmode==0: out row-major [NROWS, D]. vmode==1: out transposed per batch [B, D, S].

__global__ void __launch_bounds__(128)
qkv_gemm_kernel(const _Float16* __restrict__ xh,
                const _Float16* __restrict__ wt,
                const float*    __restrict__ bias,
                _Float16*       __restrict__ out,
                int vmode) {
    __shared__ _Float16 lds[2 * CHUNKROWS * ROWPAD];
    int lane = threadIdx.x & 31;
    int wave = threadIdx.x >> 5;
    int m0 = (blockIdx.x * 4 + wave) * 16;   // 512 M-tiles
    int n0 = blockIdx.y * 128;               // 8 N-chunks

    v8f acc[8];
#pragma unroll
    for (int j = 0; j < 8; ++j)
#pragma unroll
        for (int i = 0; i < 8; ++i) acc[j][i] = 0.0f;

    gemm_core(xh + (size_t)m0 * EMBED, EMBED,
              wt + (size_t)n0 * EMBED, EMBED, EMBED, lds, acc, lane);

    int n  = lane & 15;
    int mb = (lane < 16) ? 0 : 8;
#pragma unroll
    for (int j = 0; j < 8; ++j) {
        int e = n0 + j * 16 + n;
        float bs = bias[e];
#pragma unroll
        for (int r = 0; r < 8; ++r) {
            int m = m0 + r + mb;
            float val = acc[j][r] + bs;
            if (vmode == 0) {
                out[(size_t)m * EMBED + e] = (_Float16)val;
            } else {
                int bb = m >> 11;      // m / SEQ
                int s  = m & 2047;     // m % SEQ
                out[((size_t)bb * EMBED + e) * SEQ + s] = (_Float16)val;
            }
        }
    }
}

// ---- Pass 3: scores = (Q K^T) / sqrt(D), f16 -------------------------------

__global__ void __launch_bounds__(128)
scores_kernel(const _Float16* __restrict__ qh,
              const _Float16* __restrict__ kh,
              _Float16*       __restrict__ sc) {
    __shared__ _Float16 lds[2 * CHUNKROWS * ROWPAD];
    int lane = threadIdx.x & 31;
    int wave = threadIdx.x >> 5;
    int b  = blockIdx.z;
    int m0 = (blockIdx.x * 4 + wave) * 16;   // 128 M-tiles
    int n0 = blockIdx.y * 128;               // 16 N-chunks

    const _Float16* qb = qh + (size_t)b * SEQ * EMBED;
    const _Float16* kb = kh + (size_t)b * SEQ * EMBED;

    v8f acc[8];
#pragma unroll
    for (int j = 0; j < 8; ++j)
#pragma unroll
        for (int i = 0; i < 8; ++i) acc[j][i] = 0.0f;

    gemm_core(qb + (size_t)m0 * EMBED, EMBED,
              kb + (size_t)n0 * EMBED, EMBED, EMBED, lds, acc, lane);

    const float scale = 0.03125f;  // 1/sqrt(1024)
    int n  = lane & 15;
    int mb = (lane < 16) ? 0 : 8;
#pragma unroll
    for (int j = 0; j < 8; ++j) {
        int col = n0 + j * 16 + n;
#pragma unroll
        for (int r = 0; r < 8; ++r) {
            int m = m0 + r + mb;
            sc[((size_t)b * SEQ + m) * SEQ + col] = (_Float16)(acc[j][r] * scale);
        }
    }
}

// ---- Pass 4: row softmax, in place on f16 scores ---------------------------

__global__ void softmax_kernel(_Float16* __restrict__ sc) {
    __shared__ float red[256];
    int row = blockIdx.x;                 // 0..NROWS-1  (b*SEQ + query)
    _Float16* p = sc + (size_t)row * SEQ;
    int tid = threadIdx.x;

    float v[8];
    float mx = -3.402823466e38f;
#pragma unroll
    for (int i = 0; i < 8; ++i) {
        v[i] = (float)p[i * 256 + tid];
        mx = fmaxf(mx, v[i]);
    }
    red[tid] = mx;
    __syncthreads();
    for (int s = 128; s > 0; s >>= 1) {
        if (tid < s) red[tid] = fmaxf(red[tid], red[tid + s]);
        __syncthreads();
    }
    mx = red[0];
    __syncthreads();

    float sum = 0.0f;
#pragma unroll
    for (int i = 0; i < 8; ++i) {
        v[i] = __expf(v[i] - mx);
        sum += v[i];
    }
    red[tid] = sum;
    __syncthreads();
    for (int s = 128; s > 0; s >>= 1) {
        if (tid < s) red[tid] += red[tid + s];
        __syncthreads();
    }
    float inv = 1.0f / red[0];
#pragma unroll
    for (int i = 0; i < 8; ++i) p[i * 256 + tid] = (_Float16)(v[i] * inv);
}

// ---- Pass 5: out = P @ V (V pre-transposed [B, D, S]) ----------------------

__global__ void __launch_bounds__(128)
av_gemm_kernel(const _Float16* __restrict__ sc,
               const _Float16* __restrict__ vt,
               float*          __restrict__ out) {
    __shared__ _Float16 lds[2 * CHUNKROWS * ROWPAD];
    int lane = threadIdx.x & 31;
    int wave = threadIdx.x >> 5;
    int b  = blockIdx.z;
    int m0 = (blockIdx.x * 4 + wave) * 16;   // 128 M-tiles
    int e0 = blockIdx.y * 128;               // 8 E-chunks

    const _Float16* pb = sc + (size_t)b * SEQ * SEQ;
    const _Float16* vb = vt + (size_t)b * EMBED * SEQ;

    v8f acc[8];
#pragma unroll
    for (int j = 0; j < 8; ++j)
#pragma unroll
        for (int i = 0; i < 8; ++i) acc[j][i] = 0.0f;

    gemm_core(pb + (size_t)m0 * SEQ, SEQ,
              vb + (size_t)e0 * SEQ, SEQ, SEQ, lds, acc, lane);

    int n  = lane & 15;
    int mb = (lane < 16) ? 0 : 8;
#pragma unroll
    for (int j = 0; j < 8; ++j) {
        int e = e0 + j * 16 + n;
#pragma unroll
        for (int r = 0; r < 8; ++r) {
            int m = m0 + r + mb;
            out[((size_t)b * SEQ + m) * EMBED + e] = acc[j][r];
        }
    }
}

// ---- Launcher --------------------------------------------------------------

extern "C" void kernel_launch(void* const* d_in, const int* in_sizes, int n_in,
                              void* d_out, int out_size, void* d_ws, size_t ws_size,
                              hipStream_t stream) {
    (void)in_sizes; (void)n_in; (void)out_size; (void)ws_size;

    const float* x  = (const float*)d_in[0];
    const float* Wq = (const float*)d_in[1];
    const float* bq = (const float*)d_in[2];
    const float* Wk = (const float*)d_in[3];
    const float* bk = (const float*)d_in[4];
    const float* Wv = (const float*)d_in[5];
    const float* bv = (const float*)d_in[6];

    char* ws = (char*)d_ws;
    size_t off = 0;
    _Float16* xh  = (_Float16*)(ws + off); off += (size_t)NROWS * EMBED * 2;     // 16 MB
    _Float16* wqt = (_Float16*)(ws + off); off += (size_t)EMBED * EMBED * 2;     //  2 MB
    _Float16* wkt = (_Float16*)(ws + off); off += (size_t)EMBED * EMBED * 2;     //  2 MB
    _Float16* wvt = (_Float16*)(ws + off); off += (size_t)EMBED * EMBED * 2;     //  2 MB
    _Float16* qh  = (_Float16*)(ws + off); off += (size_t)NROWS * EMBED * 2;     // 16 MB
    _Float16* kh  = (_Float16*)(ws + off); off += (size_t)NROWS * EMBED * 2;     // 16 MB
    _Float16* vt  = (_Float16*)(ws + off); off += (size_t)NROWS * EMBED * 2;     // 16 MB [B,D,S]
    _Float16* sc  = (_Float16*)(ws + off); off += (size_t)BATCH * SEQ * SEQ * 2; // 32 MB

    // Pass 1: conversions
    cvt_x_kernel<<<(NROWS * EMBED) / 256, 256, 0, stream>>>(x, xh, NROWS * EMBED);
    transpose_w_kernel<<<(EMBED * EMBED) / 256, 256, 0, stream>>>(Wq, wqt);
    transpose_w_kernel<<<(EMBED * EMBED) / 256, 256, 0, stream>>>(Wk, wkt);
    transpose_w_kernel<<<(EMBED * EMBED) / 256, 256, 0, stream>>>(Wv, wvt);

    // Pass 2: QKV projections (512 M-tiles / 4 waves per block = 128 blocks)
    dim3 gq(128, 8, 1);
    qkv_gemm_kernel<<<gq, 128, 0, stream>>>(xh, wqt, bq, qh, 0);
    qkv_gemm_kernel<<<gq, 128, 0, stream>>>(xh, wkt, bk, kh, 0);
    qkv_gemm_kernel<<<gq, 128, 0, stream>>>(xh, wvt, bv, vt, 1);  // transposed store

    // Pass 3: scores
    scores_kernel<<<dim3(32, 16, BATCH), 128, 0, stream>>>(qh, kh, sc);

    // Pass 4: softmax (one block per query row)
    softmax_kernel<<<NROWS, 256, 0, stream>>>(sc);

    // Pass 5: context = P @ V
    av_gemm_kernel<<<dim3(32, 8, BATCH), 128, 0, stream>>>(sc, vt, (float*)d_out);
}
